// Tisa_36361193128706
// MI455X (gfx1250) — matrix operand order repaired and checked by reading, not compiled, not run on hardware
//
#include <hip/hip_runtime.h>
#include <math.h>

// ---------------- problem constants (match reference) ----------------
#define H_    16
#define KK_   8
#define L_    1024
#define D_    64
#define BM_   64     // query rows per block (16 per wave x 4 waves)
#define BN_   64     // keys per iteration
#define NW_   4      // waves per block
#define PAD_  72     // LDS row stride in bf16 elems (64 + 8 pad -> conflict-free)

typedef __attribute__((ext_vector_type(16))) __bf16 bf16x16;
typedef __attribute__((ext_vector_type(8)))  __bf16 bf16x8;
typedef __attribute__((ext_vector_type(4)))  __bf16 bf16x4;
typedef __attribute__((ext_vector_type(8)))  float  f32x8;

// Build a 16-elem bf16 fragment from two contiguous 8-elem (16B) LDS chunks.
__device__ __forceinline__ bf16x16 frag_from(const __bf16* p0, const __bf16* p1) {
  bf16x8 lo = *(const bf16x8*)p0;   // ds_load_b128
  bf16x8 hi = *(const bf16x8*)p1;   // ds_load_b128
  return __builtin_shufflevector(lo, hi,
      0,1,2,3,4,5,6,7,8,9,10,11,12,13,14,15);
}

// ---------------- kernel 1: rel_bias[H][2L-1] precompute ----------------
__global__ void Tisa_bias_precompute(const float* __restrict__ off,
                                     const float* __restrict__ amp,
                                     const float* __restrict__ shp,
                                     const float* __restrict__ bias_param,
                                     float* __restrict__ rel_bias) {
  const int idx = blockIdx.x * blockDim.x + threadIdx.x;
  const int total = H_ * (2 * L_ - 1);
  if (idx >= total) return;
  const int h = idx / (2 * L_ - 1);
  const int r = idx % (2 * L_ - 1);
  const float rel = (float)(r - (L_ - 1));

  // ALiBi-style slopes: first=-ln(1e-8)/4, last=-ln(1e-8)/256, 14 aware heads
  const float first = 4.6051702f;
  const float last  = 0.071955784f;
  float slope = 0.0f;
  if (h < H_ - 2)
    slope = first * __powf(last / first, (float)h / (float)(H_ - 2 - 1));

  float ksum = 0.0f;
#pragma unroll
  for (int kk = 0; kk < KK_; ++kk) {
    const float a = amp[kk * H_ + h];
    const float o = off[kk * H_ + h];
    const float s = shp[kk * H_ + h];
    const float sgn = (a > 0.0f) ? 1.0f : ((a < 0.0f) ? -1.0f : 0.0f);
    const float x = sgn * (rel - o) / s;
    const float sig = 1.0f / (1.0f + __expf(-x));
    ksum += fabsf(a) * sig;
  }
  const float br = __logf(ksum + 1e-8f + bias_param[h]);
  rel_bias[idx] = br - fabsf(rel) * slope;
}

// ---------------- kernel 2: bf16-WMMA flash attention ----------------
__global__ __launch_bounds__(NW_ * 32)
void Tisa_flash_attn(const float* __restrict__ q,
                     const float* __restrict__ k,
                     const float* __restrict__ v,
                     const float* __restrict__ rel_bias,
                     float* __restrict__ out) {
  __shared__ __bf16 lds_k [BN_][PAD_];       // K tile, row-major [key][d]
  __shared__ __bf16 lds_vt[D_ ][PAD_];       // V tile, transposed [d][key]
  __shared__ __bf16 lds_p [NW_][16][PAD_];   // per-wave P bounce [m][n]
  __shared__ float  lds_bias[128];           // 127-entry diagonal bias slice

  const int tid  = threadIdx.x;
  const int lane = tid & 31;
  const int wave = tid >> 5;
  const int ln   = lane & 15;       // column / row index inside fragment
  const int hh   = lane >> 4;       // half (ISA lane-split)

  const int bh = blockIdx.y;        // fused (b,h)
  const int h  = bh & (H_ - 1);
  const int m0 = blockIdx.x * BM_;

  const size_t base = (size_t)bh * L_ * D_;
  const float* qb = q + base;
  const float* kb = k + base;
  const float* vb = v + base;
  const float* biasb = rel_bias + (size_t)h * (2 * L_ - 1);

  // ---- load + prescale Q fragments (A-layout: lane=m, chunks 8h / 16+8h) ----
  const float scale = 0.125f;  // 1/sqrt(64)
  bf16x16 aq[2];
  {
    const float* qrow = qb + (size_t)(m0 + wave * 16 + ln) * D_;
#pragma unroll
    for (int ks = 0; ks < 2; ++ks) {
      bf16x16 a;
#pragma unroll
      for (int t = 0; t < 8; ++t)
        a[t]     = (__bf16)(qrow[ks * 32 + 8 * hh + t] * scale);
#pragma unroll
      for (int t = 0; t < 8; ++t)
        a[8 + t] = (__bf16)(qrow[ks * 32 + 16 + 8 * hh + t] * scale);
      aq[ks] = a;
    }
  }

  f32x8 o_acc[4] = {};              // O accumulator, C-layout, 4 D-tiles
  float m_row[8], l_row[8];
#pragma unroll
  for (int r = 0; r < 8; ++r) { m_row[r] = -INFINITY; l_row[r] = 0.0f; }

  for (int n0 = 0; n0 < L_; n0 += BN_) {
    // ---- cooperative stage: K (row-major bf16) + V (transposed bf16) ----
    {
      const int row = tid >> 1;              // key index 0..63
      const int c0  = (tid & 1) * 32;        // half-row of D
      const float* krow = kb + (size_t)(n0 + row) * D_ + c0;
      const float* vrow = vb + (size_t)(n0 + row) * D_ + c0;
#pragma unroll
      for (int j = 0; j < 8; ++j) {
        const float4 kf = *(const float4*)(krow + 4 * j);
        bf16x4 kb4 = { (__bf16)kf.x, (__bf16)kf.y, (__bf16)kf.z, (__bf16)kf.w };
        *(bf16x4*)&lds_k[row][c0 + 4 * j] = kb4;               // ds_store_b64
        const float4 vf = *(const float4*)(vrow + 4 * j);
        lds_vt[c0 + 4 * j + 0][row] = (__bf16)vf.x;            // transpose
        lds_vt[c0 + 4 * j + 1][row] = (__bf16)vf.y;
        lds_vt[c0 + 4 * j + 2][row] = (__bf16)vf.z;
        lds_vt[c0 + 4 * j + 3][row] = (__bf16)vf.w;
      }
      if (tid < 127)
        lds_bias[tid] = biasb[tid - 63 + n0 - m0 + (L_ - 1)];
    }
    __syncthreads();

    // ---- S = (Q*scale) K^T + bias,  fp32 C-layout ----
    float s[4][8];
#pragma unroll
    for (int nt = 0; nt < 4; ++nt) {
      f32x8 c = {};
#pragma unroll
      for (int ks = 0; ks < 2; ++ks) {
        // B-fragment (K^T): lane=n -> key nt*16+ln; D chunks 8h / 16+8h
        const __bf16* p = &lds_k[nt * 16 + ln][ks * 32 + 8 * hh];
        bf16x16 bfrag = frag_from(p, p + 16);
        c = __builtin_amdgcn_wmma_f32_16x16x32_bf16(
                false, aq[ks], false, bfrag, (short)0, c, false, false);
      }
#pragma unroll
      for (int r = 0; r < 8; ++r) {
        const int idx = nt * 16 + ln - (wave * 16 + r + 8 * hh) + 63;
        s[nt][r] = c[r] + lds_bias[idx];
      }
    }

    // ---- online softmax (rows live in 16-lane halves) ----
    float alpha[8];
#pragma unroll
    for (int r = 0; r < 8; ++r) {
      float tmax = fmaxf(fmaxf(s[0][r], s[1][r]), fmaxf(s[2][r], s[3][r]));
#pragma unroll
      for (int msk = 8; msk >= 1; msk >>= 1)
        tmax = fmaxf(tmax, __shfl_xor(tmax, msk, 32));
      const float nm = fmaxf(m_row[r], tmax);
      alpha[r] = __expf(m_row[r] - nm);
      m_row[r] = nm;
    }
    float tsum[8];
#pragma unroll
    for (int r = 0; r < 8; ++r) tsum[r] = 0.0f;
#pragma unroll
    for (int nt = 0; nt < 4; ++nt) {
#pragma unroll
      for (int r = 0; r < 8; ++r) {
        const float p = __expf(s[nt][r] - m_row[r]);
        tsum[r] += p;
        lds_p[wave][r + 8 * hh][nt * 16 + ln] = (__bf16)p;  // C->A bounce
      }
    }
#pragma unroll
    for (int r = 0; r < 8; ++r) {
      float t = tsum[r];
#pragma unroll
      for (int msk = 8; msk >= 1; msk >>= 1)
        t += __shfl_xor(t, msk, 32);
      l_row[r] = alpha[r] * l_row[r] + t;
    }
#pragma unroll
    for (int dt = 0; dt < 4; ++dt)
#pragma unroll
      for (int r = 0; r < 8; ++r) o_acc[dt][r] *= alpha[r];

    // ---- reload P as A-fragments (wave-local LDS; DS is in-order) ----
    bf16x16 ap[2];
#pragma unroll
    for (int ks = 0; ks < 2; ++ks) {
      const __bf16* p0 = &lds_p[wave][ln][ks * 32 + 8 * hh];
      const __bf16* p1 = &lds_p[wave][ln][ks * 32 + 16 + 8 * hh];
      ap[ks] = frag_from(p0, p1);
    }

    // ---- O += P V (V transposed in LDS -> contiguous B-fragments) ----
#pragma unroll
    for (int dt = 0; dt < 4; ++dt) {
      f32x8 c = o_acc[dt];
#pragma unroll
      for (int ks = 0; ks < 2; ++ks) {
        const __bf16* p = &lds_vt[dt * 16 + ln][ks * 32 + 8 * hh];
        bf16x16 bfrag = frag_from(p, p + 16);
        c = __builtin_amdgcn_wmma_f32_16x16x32_bf16(
                false, ap[ks], false, bfrag, (short)0, c, false, false);
      }
      o_acc[dt] = c;
    }
    __syncthreads();   // protect lds_k/lds_vt/lds_bias reuse
  }

  // ---- normalize + write out (fp32) ----
#pragma unroll
  for (int r = 0; r < 8; ++r) {
    const float inv = 1.0f / l_row[r];
    float* orow = out + base + (size_t)(m0 + wave * 16 + r + 8 * hh) * D_;
#pragma unroll
    for (int dt = 0; dt < 4; ++dt)
      orow[dt * 16 + ln] = o_acc[dt][r] * inv;
  }
}

// ---------------- launcher ----------------
extern "C" void kernel_launch(void* const* d_in, const int* in_sizes, int n_in,
                              void* d_out, int out_size, void* d_ws, size_t ws_size,
                              hipStream_t stream) {
  const float* q   = (const float*)d_in[0];
  const float* k   = (const float*)d_in[1];
  const float* v   = (const float*)d_in[2];
  const float* off = (const float*)d_in[3];
  const float* amp = (const float*)d_in[4];
  const float* shp = (const float*)d_in[5];
  const float* bp  = (const float*)d_in[6];
  float* rel_bias  = (float*)d_ws;           // H*(2L-1) floats = 131 KB
  float* out       = (float*)d_out;

  const int total = H_ * (2 * L_ - 1);
  hipLaunchKernelGGL(Tisa_bias_precompute,
                     dim3((total + 255) / 256), dim3(256), 0, stream,
                     off, amp, shp, bp, rel_bias);

  hipLaunchKernelGGL(Tisa_flash_attn,
                     dim3(L_ / BM_, 4 * H_), dim3(NW_ * 32), 0, stream,
                     q, k, v, rel_bias, out);
}
